// Qwen3Attention_14181982012145
// MI455X (gfx1250) — compile-verified, matrix operand order checked by reading, not run on hardware
//
#include <hip/hip_runtime.h>
#include <hip/hip_bf16.h>
#include <math.h>
#include <stdint.h>

// ---------------------------------------------------------------------------
// Qwen3 attention layer for MI455X (gfx1250, wave32, WMMA).
// T=3072, H=2048, NH=16, NKV=8, HD=128, GROUP=2, causal, RMSNorm+RoPE on q/k.
// Heavy math on v_wmma_f32_16x16x32_bf16 (f32 accumulate); contiguous tile
// staging on the async direct-to-LDS pipe (ASYNCcnt).
// ---------------------------------------------------------------------------

typedef __bf16 bf16;
typedef __bf16 v16bf __attribute__((ext_vector_type(16)));
typedef __bf16 v8bf  __attribute__((ext_vector_type(8)));
typedef __bf16 v4bf  __attribute__((ext_vector_type(4)));
typedef float  v8f   __attribute__((ext_vector_type(8)));

static constexpr int kT    = 3072;
static constexpr int kH    = 2048;
static constexpr int kNH   = 16;
static constexpr int kNKV  = 8;
static constexpr int kHD   = 128;
static constexpr int kQKVN = (kNH + 2 * kNKV) * kHD;   // 4096
static constexpr float kScale = 0.08838834764831845f;  // 128^-0.5
static constexpr float kLnTheta = 13.815510557964274f; // ln(1e6)

// ---- CDNA5 async global->LDS staging (GLOBAL_LOAD_ASYNC_TO_LDS_B128) ------
// Probe-confirmed signature: (int4 addrspace(1)*, int4 addrspace(3)*, imm, imm)
#if defined(__AMDGCN__) && __has_builtin(__builtin_amdgcn_global_load_async_to_lds_b128)
#define USE_ASYNC_COPY 1
typedef int b128_t __attribute__((vector_size(16)));
typedef __attribute__((address_space(1))) b128_t* gaddr_b128;
typedef __attribute__((address_space(3))) b128_t* laddr_b128;
__device__ __forceinline__ void async_copy_b128(const void* gsrc, void* ldst) {
  __builtin_amdgcn_global_load_async_to_lds_b128(
      (gaddr_b128)(uintptr_t)gsrc, (laddr_b128)(uintptr_t)ldst,
      /*imm offset=*/0, /*cpol=*/0);
}
__device__ __forceinline__ void async_wait_all() {
#if __has_builtin(__builtin_amdgcn_s_wait_asynccnt)
  __builtin_amdgcn_s_wait_asynccnt(0);
#else
  asm volatile("s_wait_asynccnt 0" ::: "memory");
#endif
}
#else
#define USE_ASYNC_COPY 0
__device__ __forceinline__ void async_wait_all() {}
#endif

// A-fragment (and mirrored B-fragment) loader for 16-bit WMMA operands.
// p32 points at a 32-element (K-chunk) row; kh = lane>>4 selects K-half group.
// Per CDNA5 ISA 7.12.2: lanes 0-15 VGPR0..3 hold K=0..7, VGPR4..7 hold K=16..23;
// lanes 16-31 hold K=8..15 and K=24..31.  => two contiguous 8xbf16 (16B) runs.
__device__ __forceinline__ v16bf load_frag(const bf16* p32, int kh) {
  const v8bf lo = *(const v8bf*)(p32 + 8 * kh);
  const v8bf hi = *(const v8bf*)(p32 + 16 + 8 * kh);
  v16bf f;
#pragma unroll
  for (int i = 0; i < 8; ++i) { f[i] = lo[i]; f[i + 8] = hi[i]; }
  return f;
}

__device__ __forceinline__ v8f bf16_wmma(v16bf a, v16bf b, v8f c) {
  // 8 args: (neg_a, A, neg_b, B, c_mod, C, reuse_a, reuse_b)
  return __builtin_amdgcn_wmma_f32_16x16x32_bf16(false, a, false, b,
                                                 (short)0, c, false, false);
}

// ---------------------------------------------------------------------------
// f32 -> bf16 bulk convert (pure bandwidth; feeds the WMMA pipelines)
// ---------------------------------------------------------------------------
__global__ __launch_bounds__(256) void cvt_bf16(const float* __restrict__ in,
                                                bf16* __restrict__ out, int n) {
  int i = (blockIdx.x * 256 + threadIdx.x) * 4;
  if (i < n) {
    float4 v = *(const float4*)(in + i);
    v4bf o;
    o[0] = (bf16)v.x; o[1] = (bf16)v.y; o[2] = (bf16)v.z; o[3] = (bf16)v.w;
    *(v4bf*)(out + i) = o;
  }
}

// ---------------------------------------------------------------------------
// QKV GEMM (3072x2048 @ 2048x4096) fused with RMSNorm + RoPE (+Q pre-scale).
// Block tile 128x128, BK=32, 256 threads = 8 waves, each wave 32x64 (2x4 WMMA).
// BN=128 == HD, so every column block is exactly one q/k/v head slice, making
// the row-wise RMSNorm reduction and the RoPE half-dim pairing block-local.
// ---------------------------------------------------------------------------
__global__ __launch_bounds__(256) void qkv_gemm_norm_rope(
    const bf16* __restrict__ A, const bf16* __restrict__ B,
    const int* __restrict__ positions,
    const float* __restrict__ qnw, const float* __restrict__ knw,
    bf16* __restrict__ qo, bf16* __restrict__ ko, bf16* __restrict__ vo) {
  __shared__ bf16 Al[128][32];
  __shared__ bf16 Bl[128][32];     // stored transposed: Bl[n][k]
  __shared__ float Cl[128][128];

  const int tid = threadIdx.x;
  const int lane = tid & 31, wave = tid >> 5;
  const int lane16 = lane & 15, kh = lane >> 4;
  const int wm = wave >> 1, wn = wave & 1;
  const int bn = blockIdx.x, bm = blockIdx.y;
  const size_t n0 = (size_t)bn * 128;
  const size_t m0 = (size_t)bm * 128;

  v8f acc[2][4] = {};

  for (int k0 = 0; k0 < kH; k0 += 32) {
    __syncthreads();
    {
      // A tile: 128 rows x 32 k, 16 elems/thread, two 16B async transfers.
      const int m = tid >> 1, kk = (tid & 1) * 16;
      const bf16* ap = A + (m0 + m) * kH + k0 + kk;
#if USE_ASYNC_COPY
      async_copy_b128(ap, &Al[m][kk]);          // bypasses VGPRs, ASYNCcnt
      async_copy_b128(ap + 8, &Al[m][kk + 8]);
#else
      *(v8bf*)&Al[m][kk] = *(const v8bf*)(ap);
      *(v8bf*)&Al[m][kk + 8] = *(const v8bf*)(ap + 8);
#endif
      // B tile: 32 k-rows x 128 n, transpose into Bl[n][k] (VGPR path).
      const int kb = tid >> 3, nb = (tid & 7) * 16;
      const bf16* bp = B + (size_t)(k0 + kb) * kQKVN + n0 + nb;
      v8bf b0 = *(const v8bf*)(bp);
      v8bf b1 = *(const v8bf*)(bp + 8);
#pragma unroll
      for (int i = 0; i < 8; ++i) {
        Bl[nb + i][kb] = b0[i];
        Bl[nb + 8 + i][kb] = b1[i];
      }
      if (k0 + 32 < kH) {  // prefetch next K-step tiles toward WGP cache
        __builtin_prefetch(ap + 32, 0, 3);
        __builtin_prefetch(bp + (size_t)32 * kQKVN, 0, 3);
      }
    }
    async_wait_all();
    __syncthreads();
    // Hoist all fragments so 12 DS loads are in flight before the WMMA burst.
    v16bf af[2], bfr[4];
#pragma unroll
    for (int tm = 0; tm < 2; ++tm)
      af[tm] = load_frag(&Al[wm * 32 + tm * 16 + lane16][0], kh);
#pragma unroll
    for (int tn = 0; tn < 4; ++tn)
      bfr[tn] = load_frag(&Bl[wn * 64 + tn * 16 + lane16][0], kh);
#pragma unroll
    for (int tn = 0; tn < 4; ++tn)
#pragma unroll
      for (int tm = 0; tm < 2; ++tm)
        acc[tm][tn] = bf16_wmma(af[tm], bfr[tn], acc[tm][tn]);
  }

  // Spill accumulators to LDS f32 tile (C layout: lane = N, vgpr + 8*kh = M)
  __syncthreads();
#pragma unroll
  for (int tm = 0; tm < 2; ++tm)
#pragma unroll
    for (int tn = 0; tn < 4; ++tn)
#pragma unroll
      for (int r = 0; r < 8; ++r)
        Cl[wm * 32 + tm * 16 + r + 8 * kh][wn * 64 + tn * 16 + lane16] =
            acc[tm][tn][r];
  __syncthreads();

  // Fused epilogue: one thread per output row
  if (tid < 128) {
    const int row = tid;
    const int grow = bm * 128 + row;
    if (bn < 24) {  // q (bn<16) or k (16<=bn<24): RMSNorm + RoPE
      const float* nw = (bn < 16) ? qnw : knw;
      float ss = 0.f;
#pragma unroll 8
      for (int c = 0; c < 128; ++c) {
        float x = Cl[row][c];
        ss += x * x;
      }
      const float rinv = rsqrtf(ss * (1.0f / 128.0f) + 1e-6f);
      const float p = (float)positions[grow];
      bf16* dst;
      float mul;
      if (bn < 16) {
        dst = qo + ((size_t)grow * kNH + bn) * kHD;
        mul = kScale;  // fold softmax 1/sqrt(d) into Q
      } else {
        dst = ko + ((size_t)grow * kNKV + (bn - 16)) * kHD;
        mul = 1.0f;
      }
      for (int c = 0; c < 64; ++c) {
        float x1 = Cl[row][c] * rinv * nw[c];
        float x2 = Cl[row][c + 64] * rinv * nw[c + 64];
        float fr = p * __expf(-(float)c * (kLnTheta / 64.0f));
        float sn, cs;
        __sincosf(fr, &sn, &cs);
        dst[c] = (bf16)((x1 * cs - x2 * sn) * mul);
        dst[c + 64] = (bf16)((x2 * cs + x1 * sn) * mul);
      }
    } else {  // v: plain copy
      bf16* dst = vo + ((size_t)grow * kNKV + (bn - 24)) * kHD;
      for (int c = 0; c < 128; ++c) dst[c] = (bf16)Cl[row][c];
    }
  }
}

// ---------------------------------------------------------------------------
// Flash attention, GQA (2 q-heads per kv-head), causal.
// Grid: (T/128 q-blocks, 16 heads). 256 threads = 8 waves; each wave owns 16
// query rows, iterates over 128-key K/V tiles staged in LDS (K via async pipe,
// V transposed through VGPRs), WMMA for QK^T and PV, online softmax with
// wave32 shfl_xor reductions.
// ---------------------------------------------------------------------------
__global__ __launch_bounds__(256) void flash_attn(
    const bf16* __restrict__ qb, const bf16* __restrict__ kb,
    const bf16* __restrict__ vb, const int* __restrict__ positions,
    bf16* __restrict__ attn) {
  __shared__ bf16 Kl[128][128];       // [key][d]
  __shared__ bf16 Vt[128][128];       // [d][key]  (transposed at load)
  __shared__ bf16 Pl[8][16][128];     // per-wave P re-layout staging

  const int tid = threadIdx.x;
  const int lane = tid & 31, wave = tid >> 5;
  const int lane16 = lane & 15, kh = lane >> 4;
  const int qx = blockIdx.x;
  const int h = blockIdx.y;
  const int hk = h >> 1;
  const int qbase = qx * 128;

  // Resident Q fragments (softmax scale already folded in by producer)
  v16bf qf[4];
  {
    const bf16* qrow =
        qb + ((size_t)(qbase + wave * 16 + lane16) * kNH + h) * kHD;
#pragma unroll
    for (int kc = 0; kc < 4; ++kc) qf[kc] = load_frag(qrow + kc * 32, kh);
  }
  int posq[8];
#pragma unroll
  for (int r = 0; r < 8; ++r)
    posq[r] = positions[qbase + wave * 16 + r + 8 * kh];

  v8f O[8] = {};
  float m_run[8], l_run[8];
#pragma unroll
  for (int r = 0; r < 8; ++r) {
    m_run[r] = -__builtin_inff();
    l_run[r] = 0.f;
  }

  for (int j = 0; j <= qx; ++j) {
    const int kvbase = j * 128;
    __syncthreads();
    {  // stage K tile row-major (async pipe) and V tile transposed (VGPR path)
      const int kr = tid >> 1, hh = tid & 1;
      const bf16* kp = kb + ((size_t)(kvbase + kr) * kNKV + hk) * kHD + hh * 64;
      const bf16* vp = vb + ((size_t)(kvbase + kr) * kNKV + hk) * kHD + hh * 64;
#pragma unroll
      for (int c8 = 0; c8 < 8; ++c8) {
#if USE_ASYNC_COPY
        async_copy_b128(kp + c8 * 8, &Kl[kr][hh * 64 + c8 * 8]);
#else
        *(v8bf*)&Kl[kr][hh * 64 + c8 * 8] = *(const v8bf*)(kp + c8 * 8);
#endif
        v8bf vv = *(const v8bf*)(vp + c8 * 8);
#pragma unroll
        for (int i = 0; i < 8; ++i) Vt[hh * 64 + c8 * 8 + i][kr] = vv[i];
      }
    }
    async_wait_all();
    __syncthreads();

    // S = Q * K^T  (16 x 128 per wave)
    v8f S[8];
#pragma unroll
    for (int nt = 0; nt < 8; ++nt) {
      v8f s = {};
#pragma unroll
      for (int kc = 0; kc < 4; ++kc) {
        v16bf kf = load_frag(&Kl[nt * 16 + lane16][kc * 32], kh);
        s = bf16_wmma(qf[kc], kf, s);
      }
      if (j == qx) {  // diagonal tile: causal mask
        const int pk = positions[kvbase + nt * 16 + lane16];
#pragma unroll
        for (int r = 0; r < 8; ++r)
          if (pk > posq[r]) s[r] = -__builtin_inff();
      }
      S[nt] = s;
    }

    // Online softmax. Row r+8*kh lives in a fixed 16-lane group => xor<16.
    float alpha[8];
#pragma unroll
    for (int r = 0; r < 8; ++r) {
      float pm = S[0][r];
#pragma unroll
      for (int nt = 1; nt < 8; ++nt) pm = fmaxf(pm, S[nt][r]);
#pragma unroll
      for (int off = 1; off < 16; off <<= 1)
        pm = fmaxf(pm, __shfl_xor(pm, off, 32));
      const float mn = fmaxf(m_run[r], pm);
      alpha[r] = __expf(m_run[r] - mn);
      m_run[r] = mn;
    }
#pragma unroll
    for (int nt = 0; nt < 8; ++nt)
#pragma unroll
      for (int r = 0; r < 8; ++r) S[nt][r] = __expf(S[nt][r] - m_run[r]);
#pragma unroll
    for (int r = 0; r < 8; ++r) {
      float ps = 0.f;
#pragma unroll
      for (int nt = 0; nt < 8; ++nt) ps += S[nt][r];
#pragma unroll
      for (int off = 1; off < 16; off <<= 1) ps += __shfl_xor(ps, off, 32);
      l_run[r] = l_run[r] * alpha[r] + ps;
    }
#pragma unroll
    for (int nt = 0; nt < 8; ++nt)
#pragma unroll
      for (int r = 0; r < 8; ++r) O[nt][r] *= alpha[r];

    // Re-layout P (C layout -> A layout) via per-wave LDS region.
    // Same-wave DS ops are in-order on CDNA5, so no barrier needed here.
#pragma unroll
    for (int nt = 0; nt < 8; ++nt)
#pragma unroll
      for (int r = 0; r < 8; ++r)
        Pl[wave][r + 8 * kh][nt * 16 + lane16] = (bf16)S[nt][r];

    // O += P * V
#pragma unroll
    for (int kc = 0; kc < 4; ++kc) {
      v16bf pa = load_frag(&Pl[wave][lane16][kc * 32], kh);
#pragma unroll
      for (int nt = 0; nt < 8; ++nt) {
        v16bf vf = load_frag(&Vt[nt * 16 + lane16][kc * 32], kh);
        O[nt] = bf16_wmma(pa, vf, O[nt]);
      }
    }
  }

#pragma unroll
  for (int r = 0; r < 8; ++r) {
    const float inv = 1.0f / l_run[r];
    const int row = qbase + wave * 16 + r + 8 * kh;
    bf16* dst = attn + ((size_t)row * kNH + h) * kHD;
#pragma unroll
    for (int nt = 0; nt < 8; ++nt)
      dst[nt * 16 + lane16] = (bf16)(O[nt][r] * inv);
  }
}

// ---------------------------------------------------------------------------
// Output projection: (3072x2048 bf16) @ (2048x2048 bf16) -> f32 out.
// ---------------------------------------------------------------------------
__global__ __launch_bounds__(256) void out_gemm(const bf16* __restrict__ A,
                                                const bf16* __restrict__ B,
                                                float* __restrict__ C) {
  __shared__ bf16 Al[128][32];
  __shared__ bf16 Bl[128][32];

  const int tid = threadIdx.x;
  const int lane = tid & 31, wave = tid >> 5;
  const int lane16 = lane & 15, kh = lane >> 4;
  const int wm = wave >> 1, wn = wave & 1;
  const int bn = blockIdx.x, bm = blockIdx.y;
  const size_t n0 = (size_t)bn * 128;
  const size_t m0 = (size_t)bm * 128;

  v8f acc[2][4] = {};

  for (int k0 = 0; k0 < kH; k0 += 32) {
    __syncthreads();
    {
      const int m = tid >> 1, kk = (tid & 1) * 16;
      const bf16* ap = A + (m0 + m) * kH + k0 + kk;
#if USE_ASYNC_COPY
      async_copy_b128(ap, &Al[m][kk]);
      async_copy_b128(ap + 8, &Al[m][kk + 8]);
#else
      *(v8bf*)&Al[m][kk] = *(const v8bf*)(ap);
      *(v8bf*)&Al[m][kk + 8] = *(const v8bf*)(ap + 8);
#endif
      const int kb = tid >> 3, nb = (tid & 7) * 16;
      const bf16* bp = B + (size_t)(k0 + kb) * kH + n0 + nb;
      v8bf b0 = *(const v8bf*)(bp);
      v8bf b1 = *(const v8bf*)(bp + 8);
#pragma unroll
      for (int i = 0; i < 8; ++i) {
        Bl[nb + i][kb] = b0[i];
        Bl[nb + 8 + i][kb] = b1[i];
      }
      if (k0 + 32 < kH) {
        __builtin_prefetch(ap + 32, 0, 3);
        __builtin_prefetch(bp + (size_t)32 * kH, 0, 3);
      }
    }
    async_wait_all();
    __syncthreads();
    v16bf af[2], bfr[4];
#pragma unroll
    for (int tm = 0; tm < 2; ++tm)
      af[tm] = load_frag(&Al[wm * 32 + tm * 16 + lane16][0], kh);
#pragma unroll
    for (int tn = 0; tn < 4; ++tn)
      bfr[tn] = load_frag(&Bl[wn * 64 + tn * 16 + lane16][0], kh);
#pragma unroll
    for (int tn = 0; tn < 4; ++tn)
#pragma unroll
      for (int tm = 0; tm < 2; ++tm)
        acc[tm][tn] = bf16_wmma(af[tm], bfr[tn], acc[tm][tn]);
  }

#pragma unroll
  for (int tm = 0; tm < 2; ++tm)
#pragma unroll
    for (int tn = 0; tn < 4; ++tn)
#pragma unroll
      for (int r = 0; r < 8; ++r) {
        const int row = (int)m0 + wm * 32 + tm * 16 + r + 8 * kh;
        const int col = (int)n0 + wn * 64 + tn * 16 + lane16;
        C[(size_t)row * kH + col] = acc[tm][tn][r];
      }
}

// ---------------------------------------------------------------------------
extern "C" void kernel_launch(void* const* d_in, const int* in_sizes, int n_in,
                              void* d_out, int out_size, void* d_ws,
                              size_t ws_size, hipStream_t stream) {
  (void)in_sizes; (void)n_in; (void)out_size; (void)ws_size;
  const int* positions = (const int*)d_in[0];
  const float* hidden = (const float*)d_in[1];
  const float* w_qkv = (const float*)d_in[2];
  const float* w_o = (const float*)d_in[3];
  const float* qnw = (const float*)d_in[4];
  const float* knw = (const float*)d_in[5];
  float* out = (float*)d_out;

  // workspace carve (bf16 elements), ~76 MB total
  bf16* p = (bf16*)d_ws;
  bf16* hsb = p;    p += (size_t)kT * kH;
  bf16* wqkvb = p;  p += (size_t)kH * kQKVN;
  bf16* wob = p;    p += (size_t)kH * kH;
  bf16* qb2 = p;    p += (size_t)kT * kNH * kHD;
  bf16* kb2 = p;    p += (size_t)kT * kNKV * kHD;
  bf16* vb2 = p;    p += (size_t)kT * kNKV * kHD;
  bf16* attnb = p;  p += (size_t)kT * kNH * kHD;

  cvt_bf16<<<(kT * kH / 4) / 256, 256, 0, stream>>>(hidden, hsb, kT * kH);
  cvt_bf16<<<(kH * kQKVN / 4) / 256, 256, 0, stream>>>(w_qkv, wqkvb,
                                                       kH * kQKVN);
  cvt_bf16<<<(kH * kH / 4) / 256, 256, 0, stream>>>(w_o, wob, kH * kH);

  qkv_gemm_norm_rope<<<dim3(kQKVN / 128, kT / 128), 256, 0, stream>>>(
      hsb, wqkvb, positions, qnw, knw, qb2, kb2, vb2);

  flash_attn<<<dim3(kT / 128, kNH), 256, 0, stream>>>(qb2, kb2, vb2, positions,
                                                      attnb);

  out_gemm<<<dim3(kH / 128, kT / 128), 256, 0, stream>>>(attnb, wob, out);
}
